// AttentionOnlyDenoiser_37185826849306
// MI455X (gfx1250) — matrix operand order repaired and checked by reading, not compiled, not run on hardware
//
#include <hip/hip_runtime.h>
#include <cstdint>

// ---------------------------------------------------------------------------
// AttentionOnlyDenoiser for MI455X (gfx1250, wave32, WMMA f32_16x16x32_f16)
// - all matrix math via v_wmma_f32_16x16x32_f16, b128 fragment loads
// - LDS staging via global_load_async_to_lds_b128 (ASYNCcnt) + double buffer
// Workspace requirement: ~385 MB (offsets in kernel_launch).
// d_out = deno [4,64,256,256] f32 followed by sims [4,65536,256] f32.
// ---------------------------------------------------------------------------

typedef __attribute__((ext_vector_type(16))) _Float16 v16h;
typedef __attribute__((ext_vector_type(8)))  _Float16 v8h;
typedef __attribute__((ext_vector_type(8)))  float    v8f;

#define NPIX 65536   // H*W
#define NB   4
#define NC   64
#define NS   256

// s_wait_asynccnt <= n  (asm-issued async copies are invisible to compiler)
#define S_WAIT_ASYNC(n) asm volatile("s_wait_asynccnt " n ::: "memory")

// One 16-byte async copy global -> LDS for this lane (ASYNCcnt tracked).
// LDS operand is the low 32 bits of the generic pointer (= LDS byte offset).
__device__ __forceinline__ void async_ld_b128(void* lds, const void* gsrc) {
  unsigned l = (unsigned)(uintptr_t)lds;
  asm volatile("global_load_async_to_lds_b128 %0, %1, off"
               :: "v"(l), "v"(gsrc) : "memory");
}

__device__ __forceinline__ v8f wmma_f16(v16h a, v16h b, v8f c) {
  return __builtin_amdgcn_wmma_f32_16x16x32_f16(false, a, false, b, (short)0, c,
                                                false, false);
}

__device__ __forceinline__ v16h cat8(v8h lo, v8h hi) {
  return __builtin_shufflevector(lo, hi, 0, 1, 2, 3, 4, 5, 6, 7,
                                 8, 9, 10, 11, 12, 13, 14, 15);
}

// A fragment (16 rows x K=32, row-major tile, lda = padded stride in halfs).
__device__ __forceinline__ v16h load_a_frag(const _Float16* base, int lda) {
  int lane = threadIdx.x & 31;
  int m = lane & 15, hi = lane >> 4;
  v8h lo = *(const v8h*)(base + m * lda + hi * 16);
  v8h hh = *(const v8h*)(base + m * lda + hi * 16 + 8);
  return cat8(lo, hh);
}

// B fragment from column-major staged tile BsT[col][k] (ldb = padded stride).
__device__ __forceinline__ v16h load_b_fragT(const _Float16* base, int ldb) {
  int lane = threadIdx.x & 31;
  int n = lane & 15, kb = (lane >> 4) * 16;
  v8h lo = *(const v8h*)(base + n * ldb + kb);
  v8h hh = *(const v8h*)(base + n * ldb + kb + 8);
  return cat8(lo, hh);
}

// ---------------------------------------------------------------------------
// Weight packing: f32 -> f16, all kept in natural [out][in] (column-major-B).
// ---------------------------------------------------------------------------
__global__ void pack_weights_k(const float* __restrict__ w0, const float* __restrict__ w1,
                               const float* __restrict__ w2, const float* __restrict__ qkvw,
                               const float* __restrict__ projw,
                               _Float16* __restrict__ w0B, _Float16* __restrict__ w1B,
                               _Float16* __restrict__ w2B, _Float16* __restrict__ qkvB,
                               _Float16* __restrict__ projB) {
  int i = blockIdx.x * blockDim.x + threadIdx.x;
  if (i < 64 * 576) {                 // [cout][k = tap*64+cin]
    int cout = i / 576, k = i % 576;
    int tap = k >> 6, cin = k & 63;
    w0B[i] = (_Float16)w0[(cout * 64 + cin) * 9 + tap];
    w1B[i] = (_Float16)w1[(cout * 64 + cin) * 9 + tap];
  }
  if (i < 256 * 64) w2B[i] = (_Float16)w2[i];      // [s][c] natural
  if (i < 192 * 64) qkvB[i] = (_Float16)qkvw[i];   // [j][c] natural
  if (i < 64 * 64)  projB[i] = (_Float16)projw[i]; // [cout][j] natural
}

// x NCHW f32 -> xh [B*N, 64] f16 (pixel-major)
__global__ void xpack_k(const float* __restrict__ x, _Float16* __restrict__ xh) {
  int i = blockIdx.x * blockDim.x + threadIdx.x;
  int c = i & 63;
  int n = (i >> 6) & (NPIX - 1);
  int b = i >> 22;
  xh[i] = (_Float16)x[(((size_t)b * NC + c) << 16) + n];
}

// ---------------------------------------------------------------------------
// 3x3 'same' conv as implicit GEMM. Tile: 64 pixels x 64 cout, 8 waves,
// K=576 in 18 chunks of 32 (each chunk = one tap, half the input channels).
// Double-buffered async LDS staging; OOB taps read a zero guard page.
// ---------------------------------------------------------------------------
__global__ __launch_bounds__(256) void conv3x3_k(const _Float16* __restrict__ xin,
    const _Float16* __restrict__ wB, const float* __restrict__ bias,
    const _Float16* __restrict__ zpad,
    _Float16* __restrict__ out, int relu) {
  __shared__ _Float16 As[2 * 64 * 40];    // [buf][pixel][k] padded
  __shared__ _Float16 BsT[2 * 64 * 40];   // [buf][cout][k] padded
  int tid = threadIdx.x;
  int b = blockIdx.x >> 10;
  int n0 = (blockIdx.x & 1023) << 6;
  int w = tid >> 5, lane = tid & 31;
  int m0 = (w & 3) * 16, c0 = (w >> 2) * 32;
  v8f acc[2] = {};
  int pr = tid >> 2, cc = (tid & 3) * 8;   // 64 rows x 4 segs of 8 halfs
  int n = n0 + pr;
  int py = n >> 8, px = n & 255;
  const size_t baseIn = (size_t)b * NPIX * 64;

  auto issue = [&](int chunk, int buf) {
    int tap = chunk >> 1;
    int cin0 = (chunk & 1) * 32;
    int dy = tap / 3 - 1, dx = tap % 3 - 1;
    int hy = py + dy, wx = px + dx;
    const _Float16* srcA =
        (hy >= 0 && hy < 256 && wx >= 0 && wx < 256)
            ? xin + baseIn + (size_t)(hy * 256 + wx) * 64 + cin0 + cc
            : zpad;
    async_ld_b128(As + buf * 2560 + pr * 40 + cc, srcA);
    async_ld_b128(BsT + buf * 2560 + pr * 40 + cc,
                  wB + pr * 576 + chunk * 32 + cc);
  };

  issue(0, 0);
  for (int chunk = 0; chunk < 18; ++chunk) {
    int cur = chunk & 1;
    if (chunk + 1 < 18) {
      issue(chunk + 1, cur ^ 1);
      S_WAIT_ASYNC("2");     // loads complete in order: chunk's pair landed
    } else {
      S_WAIT_ASYNC("0");
    }
    __syncthreads();
    v16h a = load_a_frag(As + cur * 2560 + m0 * 40, 40);
#pragma unroll
    for (int t = 0; t < 2; ++t) {
      v16h bb = load_b_fragT(BsT + cur * 2560 + (c0 + t * 16) * 40, 40);
      acc[t] = wmma_f16(a, bb, acc[t]);
    }
    __syncthreads();
  }
  int hi = lane >> 4, nn = lane & 15;
#pragma unroll
  for (int t = 0; t < 2; ++t)
#pragma unroll
    for (int r = 0; r < 8; ++r) {
      int m = m0 + r + hi * 8;
      int c = c0 + t * 16 + nn;
      float v = acc[t][r] + bias[c];
      if (relu) v = fmaxf(v, 0.f);
      out[baseIn + (size_t)(n0 + m) * 64 + c] = (_Float16)v;
    }
}

// ---------------------------------------------------------------------------
// Generic GEMM: C[M,N] = A[M,K] * B^T + bias, B given as [N][K] (natural
// weight layout). Tile 64x64, 8 waves, async double-buffered staging.
// ---------------------------------------------------------------------------
__global__ __launch_bounds__(256) void gemm_f16_k(const _Float16* __restrict__ A,
    const _Float16* __restrict__ Bw, const float* __restrict__ bias,
    _Float16* __restrict__ C, int M, int N, int K, int relu) {
  __shared__ _Float16 As[2 * 64 * 40];
  __shared__ _Float16 BsT[2 * 64 * 40];
  int tid = threadIdx.x;
  int m0g = blockIdx.x << 6, n0g = blockIdx.y << 6;
  int w = tid >> 5, lane = tid & 31;
  int m0 = (w & 3) * 16, c0 = (w >> 2) * 32;
  v8f acc[2] = {};
  int pr = tid >> 2, cc = (tid & 3) * 8;

  auto issue = [&](int it, int buf) {
    int kc = it << 5;
    async_ld_b128(As + buf * 2560 + pr * 40 + cc,
                  A + (size_t)(m0g + pr) * K + kc + cc);
    async_ld_b128(BsT + buf * 2560 + pr * 40 + cc,
                  Bw + (size_t)(n0g + pr) * K + kc + cc);
  };

  int nk = K >> 5;
  issue(0, 0);
  for (int it = 0; it < nk; ++it) {
    int cur = it & 1;
    if (it + 1 < nk) {
      issue(it + 1, cur ^ 1);
      S_WAIT_ASYNC("2");
    } else {
      S_WAIT_ASYNC("0");
    }
    __syncthreads();
    v16h a = load_a_frag(As + cur * 2560 + m0 * 40, 40);
#pragma unroll
    for (int t = 0; t < 2; ++t) {
      v16h bb = load_b_fragT(BsT + cur * 2560 + (c0 + t * 16) * 40, 40);
      acc[t] = wmma_f16(a, bb, acc[t]);
    }
    __syncthreads();
  }
  int hi = lane >> 4, nn = lane & 15;
#pragma unroll
  for (int t = 0; t < 2; ++t)
#pragma unroll
    for (int r = 0; r < 8; ++r) {
      int gm = m0g + m0 + r + hi * 8;
      int gn = n0g + c0 + t * 16 + nn;
      float v = acc[t][r] + bias[gn];
      if (relu) v = fmaxf(v, 0.f);
      C[(size_t)gm * N + gn] = (_Float16)v;
    }
}

// ---------------------------------------------------------------------------
// 1x1 conv -> logits [16 pix x 256 S] -> channel softmax -> sims f32 [b][n][s]
// + simsT f16 [b][s][n] + atomic column sums. Async one-shot staging.
// dyn LDS = 2304(Hs) + 36864(Ws) + 16384(Ls) = 55552 B
// ---------------------------------------------------------------------------
__global__ __launch_bounds__(256) void logits_softmax_k(const _Float16* __restrict__ h1,
    const _Float16* __restrict__ w2B, const float* __restrict__ b2,
    float* __restrict__ simsOut, _Float16* __restrict__ simsT,
    float* __restrict__ colsum) {
  extern __shared__ char smem[];
  _Float16* Hs = (_Float16*)smem;                  // [16][72]
  _Float16* Ws = (_Float16*)(smem + 2304);         // [256][72] col-major B
  float*    Ls = (float*)(smem + 2304 + 36864);    // [16][256]
  int tid = threadIdx.x;
  int b = blockIdx.x >> 12;
  int n0 = (blockIdx.x & 4095) << 4;
  if (tid < 128) {
    int r = tid >> 3, c = (tid & 7) * 8;
    async_ld_b128(Hs + r * 72 + c, h1 + ((size_t)b * NPIX + n0 + r) * 64 + c);
  }
#pragma unroll
  for (int i = 0; i < 8; ++i) {
    int idx = tid * 8 + i * 2048;
    int r = idx >> 6, c = idx & 63;
    async_ld_b128(Ws + r * 72 + c, w2B + idx);
  }
  S_WAIT_ASYNC("0");
  __syncthreads();
  int w = tid >> 5, lane = tid & 31;
  int s0 = w * 32;
  v8f acc[2] = {};
#pragma unroll
  for (int kc = 0; kc < 2; ++kc) {
    v16h a = load_a_frag(Hs + kc * 32, 72);
#pragma unroll
    for (int t = 0; t < 2; ++t) {
      v16h bb = load_b_fragT(Ws + (s0 + t * 16) * 72 + kc * 32, 72);
      acc[t] = wmma_f16(a, bb, acc[t]);
    }
  }
  int hi = lane >> 4, nn = lane & 15;
#pragma unroll
  for (int t = 0; t < 2; ++t)
#pragma unroll
    for (int r = 0; r < 8; ++r) {
      int m = r + hi * 8;
      int s = s0 + t * 16 + nn;
      Ls[m * 256 + s] = acc[t][r] + b2[s];
    }
  __syncthreads();
  if (tid < 16) {
    int row = tid;
    float mx = -3.0e38f;
    for (int s = 0; s < 256; ++s) mx = fmaxf(mx, Ls[row * 256 + s]);
    float sum = 0.f;
    for (int s = 0; s < 256; ++s) {
      float e = __expf(Ls[row * 256 + s] - mx);
      Ls[row * 256 + s] = e;
      sum += e;
    }
    float inv = 1.f / sum;
    for (int s = 0; s < 256; ++s) Ls[row * 256 + s] *= inv;
  }
  __syncthreads();
  float cs = 0.f;
#pragma unroll
  for (int i = 0; i < 16; ++i) {
    float p = Ls[i * 256 + tid];
    simsOut[((size_t)b * NPIX + n0 + i) * 256 + tid] = p;          // [b][n][s]
    simsT[((size_t)b * 256 + tid) * NPIX + n0 + i] = (_Float16)p;  // [b][s][n]
    cs += p;
  }
  atomicAdd(&colsum[b * 256 + tid], cs);
}

// qkv [n][192] -> kT/vT [b][c][n] f16 (transposed for wide pooling B loads)
__global__ void kvtrans_k(const _Float16* __restrict__ qkvh,
                          _Float16* __restrict__ kT, _Float16* __restrict__ vT) {
  int i = blockIdx.x * blockDim.x + threadIdx.x;   // b*64*NPIX + c*NPIX + n
  int n = i & (NPIX - 1);
  int c = (i >> 16) & 63;
  int b = i >> 22;
  size_t src = ((size_t)b * NPIX + n) * 192 + c;
  kT[i] = qkvh[src + 64];
  vT[i] = qkvh[src + 128];
}

// ---------------------------------------------------------------------------
// Superpixel pooling: spk[b,s,c] += sum_n sims[b,n,s]*k[b,n,c] (and v).
// grid = B * 16(s-tiles) * 16(split-K); per-wave K ranges, LDS+atomic reduce.
// All operands read as contiguous 16-byte global loads.
// ---------------------------------------------------------------------------
__global__ __launch_bounds__(256) void pool_spkv_k(const _Float16* __restrict__ simsT,
    const _Float16* __restrict__ kT, const _Float16* __restrict__ vT,
    float* __restrict__ spkF, float* __restrict__ spvF) {
  __shared__ float red[8 * 1024];
  int tid = threadIdx.x;
  int bi = blockIdx.x;
  int b = bi >> 8;
  int s0 = ((bi >> 4) & 15) << 4;
  int nb0 = (bi & 15) << 12;               // 4096-pixel K range per workgroup
  int w = tid >> 5, lane = tid & 31;
  int m = lane & 15, hi = lane >> 4;
  int nn = lane & 15, kb = hi * 16;
  const _Float16* spT = simsT + (size_t)b * 256 * NPIX;
  const _Float16* kTb = kT + (size_t)b * 64 * NPIX;
  const _Float16* vTb = vT + (size_t)b * 64 * NPIX;
  v8f accK[4] = {};
  v8f accV[4] = {};
  const _Float16* arow = spT + (size_t)(s0 + m) * NPIX;
  for (int ch = w; ch < 128; ch += 8) {
    int noff = nb0 + ch * 32;
    v16h a = cat8(*(const v8h*)(arow + noff + hi * 16),
                  *(const v8h*)(arow + noff + hi * 16 + 8));
#pragma unroll
    for (int t = 0; t < 4; ++t) {
      int c = t * 16 + nn;
      const _Float16* kr = kTb + (size_t)c * NPIX + noff + kb;
      const _Float16* vr = vTb + (size_t)c * NPIX + noff + kb;
      v16h bk = cat8(*(const v8h*)(kr), *(const v8h*)(kr + 8));
      v16h bv = cat8(*(const v8h*)(vr), *(const v8h*)(vr + 8));
      accK[t] = wmma_f16(a, bk, accK[t]);
      accV[t] = wmma_f16(a, bv, accV[t]);
    }
  }
#pragma unroll
  for (int t = 0; t < 4; ++t)
#pragma unroll
    for (int r = 0; r < 8; ++r)
      red[w * 1024 + (r + hi * 8) * 64 + t * 16 + nn] = accK[t][r];
  __syncthreads();
  for (int e = tid; e < 1024; e += 256) {
    float s = 0.f;
#pragma unroll
    for (int ww = 0; ww < 8; ++ww) s += red[ww * 1024 + e];
    atomicAdd(&spkF[((size_t)b * 256 + s0 + (e >> 6)) * 64 + (e & 63)], s);
  }
  __syncthreads();
#pragma unroll
  for (int t = 0; t < 4; ++t)
#pragma unroll
    for (int r = 0; r < 8; ++r)
      red[w * 1024 + (r + hi * 8) * 64 + t * 16 + nn] = accV[t][r];
  __syncthreads();
  for (int e = tid; e < 1024; e += 256) {
    float s = 0.f;
#pragma unroll
    for (int ww = 0; ww < 8; ++ww) s += red[ww * 1024 + e];
    atomicAdd(&spvF[((size_t)b * 256 + s0 + (e >> 6)) * 64 + (e & 63)], s);
  }
}

// Scale pooled tokens by 1/(colsum+1e-6); emit spk [b][s][c], spvT [b][c][s].
__global__ void scale_spkv_k(const float* __restrict__ spkF, const float* __restrict__ spvF,
                             const float* __restrict__ colsum,
                             _Float16* __restrict__ spkh, _Float16* __restrict__ spvT) {
  int i = blockIdx.x * blockDim.x + threadIdx.x;  // 4*256*64
  int b = i >> 14, s = (i >> 6) & 255, c = i & 63;
  float inv = 1.f / (colsum[b * 256 + s] + 1e-6f);
  spkh[i] = (_Float16)(spkF[i] * inv);
  spvT[((size_t)b * 64 + c) * 256 + s] = (_Float16)(spvF[i] * inv);
}

// ---------------------------------------------------------------------------
// Fused attention tail: logits = q*sp_k^T*scale -> softmax -> *sp_v -> proj
// -> + residual, written straight into NCHW deno. 32 pixels per workgroup.
// dyn LDS: sK 36864 | sVT 33792 | sP 9216 | sQ 4608 | sL 32768 | sA 16896 |
//          sO 4608 | red 2048  => 140800 B total. Async one-shot staging.
// ---------------------------------------------------------------------------
__global__ __launch_bounds__(256) void attn_fused_k(const _Float16* __restrict__ qkvh,
    const _Float16* __restrict__ spkh, const _Float16* __restrict__ spvT,
    const _Float16* __restrict__ projB, const float* __restrict__ projb,
    const float* __restrict__ x, float* __restrict__ deno) {
  extern __shared__ char smem[];
  _Float16* sK  = (_Float16*)(smem);            // [256][72]  s x c (B col-major)
  _Float16* sVT = (_Float16*)(smem + 36864);    // [64][264]  c x s (B col-major)
  _Float16* sP  = (_Float16*)(smem + 70656);    // [64][72]   cout x j
  _Float16* sQ  = (_Float16*)(smem + 79872);    // [32][72]
  float*    sL  = (float*)   (smem + 84480);    // [32][256]
  _Float16* sA  = (_Float16*)(smem + 117248);   // [32][264]
  _Float16* sO  = (_Float16*)(smem + 134144);   // [32][72]
  float*    rmax = (float*)(smem + 138752);     // [256]
  float*    rsum = rmax + 256;                  // [256]
  int tid = threadIdx.x;
  int b = blockIdx.x >> 11;
  int n0 = (blockIdx.x & 2047) << 5;
#pragma unroll
  for (int i = 0; i < 8; ++i) {
    int idx = tid * 8 + i * 2048;
    {
      int r = idx >> 6, c = idx & 63;
      async_ld_b128(sK + r * 72 + c, spkh + (size_t)b * 16384 + idx);
    }
    {
      int r = idx >> 8, c = idx & 255;
      async_ld_b128(sVT + r * 264 + c, spvT + (size_t)b * 16384 + idx);
    }
  }
#pragma unroll
  for (int i = 0; i < 2; ++i) {
    int idx = tid * 8 + i * 2048;
    int r = idx >> 6, c = idx & 63;
    async_ld_b128(sP + r * 72 + c, projB + idx);
  }
  {
    int r = tid >> 3, c = (tid & 7) * 8;
    async_ld_b128(sQ + r * 72 + c,
                  qkvh + ((size_t)b * NPIX + n0 + r) * 192 + c);
  }
  S_WAIT_ASYNC("0");
  __syncthreads();
  int w = tid >> 5, lane = tid & 31;
  int hi = lane >> 4, nn = lane & 15;
  int m0 = (w & 1) * 16;
  // stage 1: logits = q @ sp_k^T * scale
  {
    int s0 = (w >> 1) * 64;
    v8f acc[4] = {};
#pragma unroll
    for (int kc = 0; kc < 2; ++kc) {
      v16h a = load_a_frag(sQ + m0 * 72 + kc * 32, 72);
#pragma unroll
      for (int t = 0; t < 4; ++t) {
        v16h bb = load_b_fragT(sK + (s0 + t * 16) * 72 + kc * 32, 72);
        acc[t] = wmma_f16(a, bb, acc[t]);
      }
    }
#pragma unroll
    for (int t = 0; t < 4; ++t)
#pragma unroll
      for (int r = 0; r < 8; ++r)
        sL[(m0 + r + hi * 8) * 256 + s0 + t * 16 + nn] = acc[t][r] * 0.125f;
  }
  __syncthreads();
  // softmax over S=256, 8 threads per pixel row
  {
    int row = tid >> 3, seg = tid & 7;
    float* Lr = sL + row * 256 + seg * 32;
    float mx = -3.0e38f;
#pragma unroll
    for (int i = 0; i < 32; ++i) mx = fmaxf(mx, Lr[i]);
    rmax[tid] = mx;
    __syncthreads();
    float mm = -3.0e38f;
#pragma unroll
    for (int j = 0; j < 8; ++j) mm = fmaxf(mm, rmax[row * 8 + j]);
    float sum = 0.f;
#pragma unroll
    for (int i = 0; i < 32; ++i) {
      float e = __expf(Lr[i] - mm);
      Lr[i] = e;
      sum += e;
    }
    rsum[tid] = sum;
    __syncthreads();
    float tot = 0.f;
#pragma unroll
    for (int j = 0; j < 8; ++j) tot += rsum[row * 8 + j];
    float inv = 1.f / tot;
    _Float16* Ar = sA + row * 264 + seg * 32;
#pragma unroll
    for (int i = 0; i < 32; ++i) Ar[i] = (_Float16)(Lr[i] * inv);
  }
  __syncthreads();
  // stage 2: out = attn @ sp_v   (K = 256)
  int c0 = (w >> 1) * 16;
  {
    v8f acc = {};
#pragma unroll
    for (int kc = 0; kc < 8; ++kc) {
      v16h a = load_a_frag(sA + m0 * 264 + kc * 32, 264);
      v16h bb = load_b_fragT(sVT + c0 * 264 + kc * 32, 264);
      acc = wmma_f16(a, bb, acc);
    }
#pragma unroll
    for (int r = 0; r < 8; ++r)
      sO[(m0 + r + hi * 8) * 72 + c0 + nn] = (_Float16)acc[r];
  }
  __syncthreads();
  // stage 3: proj + residual, write NCHW
  {
    v8f acc = {};
#pragma unroll
    for (int kc = 0; kc < 2; ++kc) {
      v16h a = load_a_frag(sO + m0 * 72 + kc * 32, 72);
      v16h bb = load_b_fragT(sP + c0 * 72 + kc * 32, 72);
      acc = wmma_f16(a, bb, acc);
    }
#pragma unroll
    for (int r = 0; r < 8; ++r) {
      int mloc = m0 + r + hi * 8;
      int c = c0 + nn;
      size_t gi = ((size_t)(b * 64 + c)) * NPIX + n0 + mloc;
      deno[gi] = acc[r] + projb[c] + x[gi];
    }
  }
}

// ---------------------------------------------------------------------------
extern "C" void kernel_launch(void* const* d_in, const int* in_sizes, int n_in,
                              void* d_out, int out_size, void* d_ws, size_t ws_size,
                              hipStream_t stream) {
  (void)in_sizes; (void)n_in; (void)out_size; (void)ws_size;
  const float* x     = (const float*)d_in[0];
  const float* w0    = (const float*)d_in[1];
  const float* b0    = (const float*)d_in[2];
  const float* w1    = (const float*)d_in[3];
  const float* b1    = (const float*)d_in[4];
  const float* w2    = (const float*)d_in[5];
  const float* b2    = (const float*)d_in[6];
  const float* qkvw  = (const float*)d_in[7];
  const float* qkvb  = (const float*)d_in[8];
  const float* projw = (const float*)d_in[9];
  const float* projb = (const float*)d_in[10];

  float* deno = (float*)d_out;
  float* sims = deno + (size_t)NB * NC * NPIX;  // second output

  char* ws = (char*)d_ws;
  _Float16* xh    = (_Float16*)(ws + 0);          // 33.5 MB [B*N,64]
  _Float16* f0h   = (_Float16*)(ws + 33554432);   // 33.5 MB
  _Float16* h1h   = (_Float16*)(ws + 67108864);   // 33.5 MB
  _Float16* qkvh  = (_Float16*)(ws + 100663296);  // 100.7 MB [B*N,192]
  _Float16* simsT = (_Float16*)(ws + 201326592);  // 134.2 MB [B,256,N]
  _Float16* kT    = (_Float16*)(ws + 335544320);  // 33.5 MB [B,64,N]
  _Float16* vT    = (_Float16*)(ws + 369098752);  // 33.5 MB [B,64,N]
  float*    colsum= (float*)   (ws + 402653184);  // 4 KB
  float*    spkF  = (float*)   (ws + 402657280);  // 256 KB [B,256,64]
  float*    spvF  = (float*)   (ws + 402919424);  // 256 KB
  _Float16* spkh  = (_Float16*)(ws + 403181568);  // 128 KB [B,256,64]
  _Float16* spvT  = (_Float16*)(ws + 403312640);  // 128 KB [B,64,256]
  _Float16* w0B   = (_Float16*)(ws + 403443712);  // [64,576]
  _Float16* w1B   = (_Float16*)(ws + 403517440);  // [64,576]
  _Float16* w2B   = (_Float16*)(ws + 403591168);  // [256,64]
  _Float16* qkvB  = (_Float16*)(ws + 403623936);  // [192,64]
  _Float16* projB = (_Float16*)(ws + 403648512);  // [64,64]
  _Float16* zpad  = (_Float16*)(ws + 403656704);  // 256 B zero guard page

  // zero the atomic accumulators (colsum | spkF | spvF contiguous) + guard
  hipMemsetAsync(colsum, 0, 4096 + 262144 + 262144, stream);
  hipMemsetAsync(zpad, 0, 256, stream);

  pack_weights_k<<<144, 256, 0, stream>>>(w0, w1, w2, qkvw, projw,
                                          w0B, w1B, w2B, qkvB, projB);
  xpack_k<<<65536, 256, 0, stream>>>(x, xh);
  conv3x3_k<<<4096, 256, 0, stream>>>(xh, w0B, b0, zpad, f0h, 0);
  conv3x3_k<<<4096, 256, 0, stream>>>(f0h, w1B, b1, zpad, h1h, 1);

  hipFuncSetAttribute(reinterpret_cast<const void*>(logits_softmax_k),
                      hipFuncAttributeMaxDynamicSharedMemorySize, 55552);
  logits_softmax_k<<<16384, 256, 55552, stream>>>(h1h, w2B, b2, sims, simsT, colsum);

  gemm_f16_k<<<dim3(4096, 3), 256, 0, stream>>>(xh, qkvB, qkvb, qkvh,
                                                NB * NPIX, 192, 64, 0);
  kvtrans_k<<<65536, 256, 0, stream>>>(qkvh, kT, vT);
  pool_spkv_k<<<1024, 256, 0, stream>>>(simsT, kT, vT, spkF, spvF);
  scale_spkv_k<<<256, 256, 0, stream>>>(spkF, spvF, colsum, spkh, spvT);

  hipFuncSetAttribute(reinterpret_cast<const void*>(attn_fused_k),
                      hipFuncAttributeMaxDynamicSharedMemorySize, 140800);
  attn_fused_k<<<8192, 256, 140800, stream>>>(qkvh, spkh, spvT, projB, projb,
                                              x, deno);
}
